// ATT_14972255993877
// MI455X (gfx1250) — compile-verified, matrix operand order checked by reading, not run on hardware
//
#include <hip/hip_runtime.h>
#include <hip/hip_bf16.h>
#include <stdint.h>

// ---- CDNA5 WMMA types ----
typedef __attribute__((ext_vector_type(16))) __bf16 v16bf;
typedef __attribute__((ext_vector_type(8)))  float  v8f;

union FragB16 { v16bf v; uint32_t u[8]; };
union FragF32 { v8f v; float f[8]; };

__device__ __forceinline__ unsigned short f2bf(float f) {
    union { float f; uint32_t u; } x; x.f = f;
    uint32_t r = x.u + 0x7FFFu + ((x.u >> 16) & 1u);   // round-to-nearest-even
    return (unsigned short)(r >> 16);
}

#define BATCH 4096
#define NKEY  128
#define UDIM  256

// ------------------------------------------------------------------
// prep: Mt = (Wq^T Wk)^T (bf16), u2 = Wk^T bq (f32), Wv/Wf -> bf16
// grid: 256 (Mt rows) + 1 (u2) + 64 (Wv) + 128 (Wf) = 449 blocks x 256
// ------------------------------------------------------------------
__global__ __launch_bounds__(256)
void att_prep(const float* __restrict__ Wq, const float* __restrict__ Wk,
              const float* __restrict__ bq, const float* __restrict__ Wv,
              const float* __restrict__ Wf,
              unsigned short* __restrict__ Mt, float* __restrict__ u2,
              unsigned short* __restrict__ Wv_bf, unsigned short* __restrict__ Wf_bf)
{
    const int bid = blockIdx.x, tid = threadIdx.x;
    if (bid < 256) {                       // Mt[l][j] = sum_i Wk[i][l]*Wq[i][j]
        const int l = bid, j = tid;
        float acc = 0.f;
        for (int i = 0; i < 256; ++i)
            acc = fmaf(Wk[i * 256 + l], Wq[i * 256 + j], acc);
        Mt[l * 256 + j] = f2bf(acc);
    } else if (bid == 256) {               // u2[l] = sum_i bq[i]*Wk[i][l]
        const int l = tid;
        float acc = 0.f;
        for (int i = 0; i < 256; ++i)
            acc = fmaf(bq[i], Wk[i * 256 + l], acc);
        u2[l] = acc;
    } else if (bid < 256 + 1 + 64) {       // Wv f32 -> bf16 (65536 elems)
        const int e = (bid - 257) * 1024 + tid * 4;
        #pragma unroll
        for (int q = 0; q < 4; ++q) Wv_bf[e + q] = f2bf(Wv[e + q]);
    } else {                               // Wf f32 -> bf16 (131072 elems)
        const int e = (bid - 321) * 1024 + tid * 4;
        #pragma unroll
        for (int q = 0; q < 4; ++q) Wf_bf[e + q] = f2bf(Wf[e + q]);
    }
}

// ------------------------------------------------------------------
// WMMA GEMM: out[r,n] = act( sum_k In[r,k]*Wt[n,k] + bias[n] )
// In f32 (optionally concat of two K/2 halves), Wt bf16 (N=256 rows, K cols).
// Block = 16 rows x 256 cols; 256 threads = 8 waves, 2 n-tiles per wave.
// ------------------------------------------------------------------
template<int K, bool CONCAT, bool RELU>
__global__ __launch_bounds__(256)
void att_gemm(const float* __restrict__ in0, const float* __restrict__ in1,
              const unsigned short* __restrict__ wt, const float* __restrict__ bias,
              float* __restrict__ out)
{
    constexpr int PITCH = K + 8;                    // bank-conflict padding
    __shared__ unsigned short ldsA[16 * PITCH];
    const int row0 = blockIdx.x * 16;
    const int tid  = threadIdx.x;

    // Stage A tile (16 x K) into LDS as bf16
    for (int e = tid; e < 16 * K; e += 256) {
        const int r = e / K, k = e % K;
        float v;
        if (CONCAT) {
            v = (k < K / 2) ? in0[(size_t)(row0 + r) * (K / 2) + k]
                            : in1[(size_t)(row0 + r) * (K / 2) + (k - K / 2)];
        } else {
            v = in0[(size_t)(row0 + r) * K + k];
        }
        ldsA[r * PITCH + k] = f2bf(v);
    }
    __syncthreads();

    const int lane  = tid & 31;
    const int wave  = tid >> 5;          // 0..7
    const int mn    = lane & 15;         // M for A-frag, N for B-frag
    const int hiOff = (lane >> 4) << 3;  // 0 or 8 (K-half select)

    v8f acc0 = {}; v8f acc1 = {};

    for (int kk = 0; kk < K; kk += 32) {
        FragB16 a;
        #pragma unroll
        for (int v = 0; v < 8; ++v) {
            const int k0 = ((v < 4) ? 2 * v : 2 * v + 8) + hiOff + kk;
            a.u[v] = *(const uint32_t*)&ldsA[mn * PITCH + k0];
        }
        FragB16 b0, b1;
        const int n0 = (wave * 2 + 0) * 16 + mn;
        const int n1 = (wave * 2 + 1) * 16 + mn;
        #pragma unroll
        for (int v = 0; v < 8; ++v) {
            const int k0 = ((v < 4) ? 2 * v : 2 * v + 8) + hiOff + kk;
            b0.u[v] = *(const uint32_t*)&wt[(size_t)n0 * K + k0];
            b1.u[v] = *(const uint32_t*)&wt[(size_t)n1 * K + k0];
        }
        acc0 = __builtin_amdgcn_wmma_f32_16x16x32_bf16(false, a.v, false, b0.v,
                                                       (short)0, acc0, false, false);
        acc1 = __builtin_amdgcn_wmma_f32_16x16x32_bf16(false, a.v, false, b1.v,
                                                       (short)0, acc1, false, false);
    }

    // Epilogue: D layout -> lane n = lane&15, M = v + 8*laneHi
    #pragma unroll
    for (int t = 0; t < 2; ++t) {
        const int n = (wave * 2 + t) * 16 + mn;
        const float bn = bias ? bias[n] : 0.f;
        FragF32 c; c.v = t ? acc1 : acc0;
        #pragma unroll
        for (int v = 0; v < 8; ++v) {
            const int m = v + hiOff;
            float val = c.f[v] + bn;
            if (RELU) val = fmaxf(val, 0.f);
            out[(size_t)(row0 + m) * 256 + n] = val;
        }
    }
}

// ------------------------------------------------------------------
// attn: per batch row b — async-DMA keys[b] (128x256 f32 = 128KB) into LDS
// via GLOBAL_LOAD_ASYNC_TO_LDS_B128 (ASYNCcnt path, no VGPR round trip),
// scores = S_b . keys_bn / 16, softmax over n, w_b = sum_n p_n keys_bn.
// 4096 blocks x 128 threads, dynamic LDS = 133120 bytes (2 WGs / WGP).
// ------------------------------------------------------------------
#define ATT_LDS_BYTES (NKEY * UDIM * 4 + 256 * 4 + 128 * 4 + 128 * 4)  // 133120

__global__ __launch_bounds__(128)
void att_softmax_ctx(const float* __restrict__ S, const float* __restrict__ keys,
                     float* __restrict__ Wout)
{
    extern __shared__ char smem[];
    float* kF  = (float*)smem;                               // [128][256] f32, n-major
    float* sS  = (float*)(smem + NKEY * UDIM * 4);           // 256 f32
    float* red = (float*)(smem + NKEY * UDIM * 4 + 1024);    // 128 f32
    float* pP  = (float*)(smem + NKEY * UDIM * 4 + 1536);    // 128 f32

    const int b   = blockIdx.x;
    const int tid = threadIdx.x;
    const float* kb = keys + (size_t)b * NKEY * UDIM;

    sS[tid]       = S[(size_t)b * 256 + tid];
    sS[tid + 128] = S[(size_t)b * 256 + tid + 128];

    // Async copy keys[b] -> LDS: 16B per lane per op, 64 ops/thread = 128KB total.
    // Generic LDS pointer's low 32 bits are the LDS byte address (ISA 10.2).
    {
        const uint64_t gbase = (uint64_t)(uintptr_t)kb;
        const uint32_t lbase = (uint32_t)(uintptr_t)kF;
        #pragma unroll 8
        for (int it = 0; it < 64; ++it) {
            const uint32_t off = ((uint32_t)it * 128u + (uint32_t)tid) * 16u;
            const uint32_t l   = lbase + off;
            asm volatile("global_load_async_to_lds_b128 %0, %1, %2"
                         :: "v"(l), "v"(off), "s"(gbase) : "memory");
        }
        asm volatile("s_wait_asynccnt 0x0" ::: "memory");
    }
    __syncthreads();

    // scores[n=tid] = (s_b . keys[b,n]) / sqrt(U); rotated index keeps
    // LDS dword-bank = (uu+tid)%64 distinct across lanes -> conflict-free.
    float sc = 0.f;
    #pragma unroll 4
    for (int uu = 0; uu < 256; ++uu) {
        const int u = (uu + tid) & 255;
        sc = fmaf(sS[u], kF[tid * 256 + u], sc);
    }
    sc *= 0.0625f;

    // softmax over 128 lanes via LDS tree reductions
    red[tid] = sc; __syncthreads();
    for (int s = 64; s > 0; s >>= 1) {
        if (tid < s) red[tid] = fmaxf(red[tid], red[tid + s]);
        __syncthreads();
    }
    const float mx = red[0]; __syncthreads();
    const float e = __expf(sc - mx);
    red[tid] = e; __syncthreads();
    for (int s = 64; s > 0; s >>= 1) {
        if (tid < s) red[tid] += red[tid + s];
        __syncthreads();
    }
    const float inv = 1.f / red[0];
    pP[tid] = e * inv; __syncthreads();

    // w[u] = sum_n p[n] * keys[b,n,u]; lane stride 4B -> conflict-free
    float w0 = 0.f, w1 = 0.f;
    for (int n = 0; n < NKEY; ++n) {
        const float p = pP[n];
        w0 = fmaf(p, kF[n * 256 + tid],       w0);
        w1 = fmaf(p, kF[n * 256 + tid + 128], w1);
    }
    Wout[(size_t)b * 256 + tid]       = w0;
    Wout[(size_t)b * 256 + tid + 128] = w1;
}

// ------------------------------------------------------------------
extern "C" void kernel_launch(void* const* d_in, const int* in_sizes, int n_in,
                              void* d_out, int out_size, void* d_ws, size_t ws_size,
                              hipStream_t stream)
{
    const float* query = (const float*)d_in[0];
    const float* keys  = (const float*)d_in[1];
    const float* Wq    = (const float*)d_in[2];
    const float* bq    = (const float*)d_in[3];
    const float* Wk    = (const float*)d_in[4];
    // d_in[5] = bk: provably cancels under softmax shift-invariance
    const float* Wv    = (const float*)d_in[6];
    const float* bv    = (const float*)d_in[7];
    const float* Wf    = (const float*)d_in[8];
    const float* bf_   = (const float*)d_in[9];

    char* ws = (char*)d_ws;
    unsigned short* Mt    = (unsigned short*)(ws + 0);          // 256*256*2 = 131072
    unsigned short* Wv_bf = (unsigned short*)(ws + 131072);     // 131072
    unsigned short* Wf_bf = (unsigned short*)(ws + 262144);     // 262144
    float*          u2    = (float*)(ws + 524288);              // 1024
    float*          Sbuf  = (float*)(ws + 525312);              // 4096*256*4 = 4 MB
    float*          Wbuf  = (float*)(ws + 4719616);             // 4 MB
    float*          Cbuf  = (float*)(ws + 8913920);             // 4 MB  (total ~12.5 MB)

    // 1) one-time weight algebra (Mt = Wk^T Wq layout for s = query@M + u2)
    att_prep<<<449, 256, 0, stream>>>(Wq, Wk, bq, Wv, Wf, Mt, u2, Wv_bf, Wf_bf);

    // 2) S = query @ Mt^T + u2            (4096x256, K=256, WMMA bf16)
    att_gemm<256, false, false><<<BATCH / 16, 256, 0, stream>>>(query, nullptr, Mt, u2, Sbuf);

    // 3) streaming softmax-attention over keys (reads keys exactly once, async->LDS)
    att_softmax_ctx<<<BATCH, 128, ATT_LDS_BYTES, stream>>>(Sbuf, keys, Wbuf);

    // 4) ctx = w @ Wv^T + bv              (4096x256, K=256, WMMA bf16)
    att_gemm<256, false, false><<<BATCH / 16, 256, 0, stream>>>(Wbuf, nullptr, Wv_bf, bv, Cbuf);

    // 5) out = relu([ctx|query] @ Wf^T + bf)   (4096x256, K=512, WMMA bf16)
    att_gemm<512, true, true><<<BATCH / 16, 256, 0, stream>>>(Cbuf, query, Wf_bf, bf_,
                                                              (float*)d_out);
}
